// PN2PartSegSsgEncoder_41532333752453
// MI455X (gfx1250) — matrix operand order repaired and checked by reading, not compiled, not run on hardware
//
#include <hip/hip_runtime.h>
#include <hip/hip_bf16.h>
#include <math.h>

typedef _Float16 h16;
typedef __attribute__((ext_vector_type(16))) _Float16 v16h;
typedef __attribute__((ext_vector_type(8)))  _Float16 v8h;
typedef __attribute__((ext_vector_type(8)))  float    v8f;

// ---------------------------------------------------------------------------
// Farthest point sampling: one workgroup per batch, dist[] lives in LDS.
// ---------------------------------------------------------------------------
__global__ void fps_kernel(const float* __restrict__ xyz, int N, int npoint,
                           int* __restrict__ out) {
  __shared__ float dist[4096];
  __shared__ float rv[256];
  __shared__ int   ri[256];
  __shared__ int   sf;
  const int b = blockIdx.x, tid = threadIdx.x;
  const float* p = xyz + (size_t)b * N * 3;
  for (int j = tid; j < N; j += blockDim.x) dist[j] = 1e10f;
  if (tid == 0) sf = 0;
  __syncthreads();
  for (int i = 0; i < npoint; ++i) {
    const int far = sf;
    if (tid == 0) out[b * npoint + i] = far;
    const float cx = p[far * 3 + 0], cy = p[far * 3 + 1], cz = p[far * 3 + 2];
    float bm = -1.0f; int bi = 0;
    for (int j = tid; j < N; j += blockDim.x) {
      const float dx = p[j * 3 + 0] - cx, dy = p[j * 3 + 1] - cy, dz = p[j * 3 + 2] - cz;
      float dm = fminf(dist[j], dx * dx + dy * dy + dz * dz);
      dist[j] = dm;
      if (dm > bm) { bm = dm; bi = j; }
    }
    rv[tid] = bm; ri[tid] = bi;
    __syncthreads();
    for (int s = 128; s > 0; s >>= 1) {
      if (tid < s) {
        if (rv[tid + s] > rv[tid] ||
            (rv[tid + s] == rv[tid] && ri[tid + s] < ri[tid])) {
          rv[tid] = rv[tid + s]; ri[tid] = ri[tid + s];
        }
      }
      __syncthreads();
    }
    if (tid == 0) sf = ri[0];
    __syncthreads();
  }
}

// ---------------------------------------------------------------------------
// Ball query: one thread per (b,s) center; sequential index scan == sorted idx.
// ---------------------------------------------------------------------------
__global__ void ball_kernel(const float* __restrict__ xyz, int N,
                            const int* __restrict__ fps_idx, int S,
                            float r2, int ns,
                            int* __restrict__ idx_out, float* __restrict__ new_xyz,
                            int total) {
  const int t = blockIdx.x * blockDim.x + threadIdx.x;
  if (t >= total) return;
  const int b = t / S;
  const float* p = xyz + (size_t)b * N * 3;
  const int fi = fps_idx[t];
  const float cx = p[fi * 3 + 0], cy = p[fi * 3 + 1], cz = p[fi * 3 + 2];
  new_xyz[t * 3 + 0] = cx; new_xyz[t * 3 + 1] = cy; new_xyz[t * 3 + 2] = cz;
  int* oi = idx_out + (size_t)t * ns;
  int cnt = 0, first = 0; bool has = false;
  for (int j = 0; j < N && cnt < ns; ++j) {
    const float dx = p[j * 3 + 0] - cx, dy = p[j * 3 + 1] - cy, dz = p[j * 3 + 2] - cz;
    if (dx * dx + dy * dy + dz * dz <= r2) {
      if (!has) { first = j; has = true; }
      oi[cnt++] = j;
    }
  }
  for (int k = cnt; k < ns; ++k) oi[k] = first;
}

// ---------------------------------------------------------------------------
// Grouping kernels -> padded f16 GEMM A matrices
// ---------------------------------------------------------------------------
__global__ void group_sa1_kernel(const float* __restrict__ xyz,
                                 const float* __restrict__ new_xyz,
                                 const int* __restrict__ idx,
                                 h16* __restrict__ A, int total,
                                 int S, int ns, int N, int Cp) {
  const int t = blockIdx.x * blockDim.x + threadIdx.x;
  if (t >= total) return;
  const int c = t % Cp, row = t / Cp;
  const int k = row % ns, bs = row / ns;
  const int b = bs / S;
  const int j = idx[(size_t)bs * ns + k];
  const float* p = xyz + ((size_t)b * N + j) * 3;
  float v = 0.0f;
  if (c < 3)      v = p[c] - new_xyz[bs * 3 + c];
  else if (c < 6) v = p[c - 3];
  A[t] = (h16)v;
}

__global__ void group_sa2_kernel(const float* __restrict__ xyz1,
                                 const h16* __restrict__ pts,
                                 const float* __restrict__ new_xyz,
                                 const int* __restrict__ idx,
                                 h16* __restrict__ A, int total,
                                 int S, int ns, int N, int Cpts, int Cp) {
  const int t = blockIdx.x * blockDim.x + threadIdx.x;
  if (t >= total) return;
  const int c = t % Cp, row = t / Cp;
  const int k = row % ns, bs = row / ns;
  const int b = bs / S;
  const int j = idx[(size_t)bs * ns + k];
  float v = 0.0f;
  if (c < 3)             v = xyz1[((size_t)b * N + j) * 3 + c] - new_xyz[bs * 3 + c];
  else if (c < 3 + Cpts) v = (float)pts[((size_t)b * N + j) * Cpts + (c - 3)];
  A[t] = (h16)v;
}

__global__ void group_all_kernel(const float* __restrict__ xyz2,
                                 const h16* __restrict__ pts,
                                 h16* __restrict__ A, int total,
                                 int Cpts, int Cp) {
  const int t = blockIdx.x * blockDim.x + threadIdx.x;
  if (t >= total) return;
  const int c = t % Cp, row = t / Cp;   // row = b*N2 + k
  float v = 0.0f;
  if (c < 3)             v = xyz2[(size_t)row * 3 + c];
  else if (c < 3 + Cpts) v = (float)pts[(size_t)row * Cpts + (c - 3)];
  A[t] = (h16)v;
}

// ---------------------------------------------------------------------------
// Max-pool over the sample axis
// ---------------------------------------------------------------------------
__global__ void maxpool_kernel(const h16* __restrict__ h, h16* __restrict__ out,
                               int total, int ns, int O) {
  const int t = blockIdx.x * blockDim.x + threadIdx.x;
  if (t >= total) return;
  const int o = t % O, bs = t / O;
  const h16* src = h + ((size_t)bs * ns) * O + o;
  float m = -3.4e38f;
  for (int k = 0; k < ns; ++k) m = fmaxf(m, (float)src[(size_t)k * O]);
  out[t] = (h16)m;
}

// ---------------------------------------------------------------------------
// 3-NN indices + normalized inverse-distance weights
// ---------------------------------------------------------------------------
__global__ void knn3_kernel(const float* __restrict__ xyz1,
                            const float* __restrict__ xyz2,
                            int total, int N1, int S2,
                            int* __restrict__ idx3, float* __restrict__ w3) {
  const int t = blockIdx.x * blockDim.x + threadIdx.x;
  if (t >= total) return;
  const int b = t / N1;
  const float* q = xyz1 + (size_t)t * 3;
  const float* p = xyz2 + (size_t)b * S2 * 3;
  const float qx = q[0], qy = q[1], qz = q[2];
  float d0 = 1e30f, d1 = 1e30f, d2 = 1e30f;
  int   i0 = 0, i1 = 0, i2 = 0;
  for (int j = 0; j < S2; ++j) {
    const float dx = p[j * 3 + 0] - qx, dy = p[j * 3 + 1] - qy, dz = p[j * 3 + 2] - qz;
    const float d = dx * dx + dy * dy + dz * dz;
    if (d < d0)      { d2 = d1; i2 = i1; d1 = d0; i1 = i0; d0 = d; i0 = j; }
    else if (d < d1) { d2 = d1; i2 = i1; d1 = d; i1 = j; }
    else if (d < d2) { d2 = d; i2 = j; }
  }
  const float w0 = 1.0f / (d0 + 1e-8f), w1 = 1.0f / (d1 + 1e-8f), w2 = 1.0f / (d2 + 1e-8f);
  const float s = w0 + w1 + w2;
  idx3[t * 3 + 0] = i0; idx3[t * 3 + 1] = i1; idx3[t * 3 + 2] = i2;
  w3[t * 3 + 0] = w0 / s; w3[t * 3 + 1] = w1 / s; w3[t * 3 + 2] = w2 / s;
}

// ---------------------------------------------------------------------------
// FP concat / interp kernels
// ---------------------------------------------------------------------------
__global__ void concat_fp3_kernel(const h16* __restrict__ p1f, const h16* __restrict__ p2f,
                                  h16* __restrict__ A, int total, int C1, int C2, int S) {
  const int t = blockIdx.x * blockDim.x + threadIdx.x;
  if (t >= total) return;
  const int Cp = C1 + C2;
  const int c = t % Cp, row = t / Cp, b = row / S;
  A[t] = (c < C1) ? p1f[(size_t)row * C1 + c] : p2f[(size_t)b * C2 + (c - C1)];
}

__global__ void interp_fp2_kernel(const h16* __restrict__ pts1, const h16* __restrict__ feat2,
                                  const int* __restrict__ idx3, const float* __restrict__ w3,
                                  h16* __restrict__ A, int total,
                                  int N1, int S2, int C1, int C2) {
  const int t = blockIdx.x * blockDim.x + threadIdx.x;
  if (t >= total) return;
  const int Cp = C1 + C2;
  const int c = t % Cp, row = t / Cp, b = row / N1;
  if (c < C1) { A[t] = pts1[(size_t)row * C1 + c]; return; }
  const int ch = c - C1;
  const int* id = idx3 + (size_t)row * 3;
  const float* w = w3 + (size_t)row * 3;
  const float v = w[0] * (float)feat2[((size_t)b * S2 + id[0]) * C2 + ch] +
                  w[1] * (float)feat2[((size_t)b * S2 + id[1]) * C2 + ch] +
                  w[2] * (float)feat2[((size_t)b * S2 + id[2]) * C2 + ch];
  A[t] = (h16)v;
}

__global__ void interp_fp1_kernel(const float* __restrict__ xyz, const h16* __restrict__ feat2,
                                  const int* __restrict__ idx3, const float* __restrict__ w3,
                                  h16* __restrict__ A, int total,
                                  int N1, int S2, int C2, int Cp) {
  const int t = blockIdx.x * blockDim.x + threadIdx.x;
  if (t >= total) return;
  const int c = t % Cp, row = t / Cp, b = row / N1;
  float v = 0.0f;
  if (c < 3)            v = xyz[(size_t)row * 3 + c];
  else if (c < 6)       v = xyz[(size_t)row * 3 + (c - 3)];
  else if (c < 6 + C2) {
    const int ch = c - 6;
    const int* id = idx3 + (size_t)row * 3;
    const float* w = w3 + (size_t)row * 3;
    v = w[0] * (float)feat2[((size_t)b * S2 + id[0]) * C2 + ch] +
        w[1] * (float)feat2[((size_t)b * S2 + id[1]) * C2 + ch] +
        w[2] * (float)feat2[((size_t)b * S2 + id[2]) * C2 + ch];
  }
  A[t] = (h16)v;
}

// ---------------------------------------------------------------------------
// Weight prep: (O,C) f32 -> WMMA-B-fragment-packed f16.
// Bpack[((tn*KB + kb)*32 + lane)*16 + j] = W[o, k], with
//   o = tn*16 + (lane&15),  k = kb*32 + (lane>>4)*16 + j   (0 if k >= C)
// so a lane's 16-element B fragment is one contiguous 32-byte load.
// ---------------------------------------------------------------------------
__global__ void pack_w_kernel(const float* __restrict__ W, h16* __restrict__ wt,
                              int C, int Kp, int O, int total) {
  const int t = blockIdx.x * blockDim.x + threadIdx.x;
  if (t >= total) return;
  const int KB   = Kp >> 5;
  const int j    = t & 15;
  const int lane = (t >> 4) & 31;
  const int kb   = (t >> 9) % KB;
  const int tn   = t / (KB << 9);
  const int k = (kb << 5) + ((lane >> 4) << 4) + j;
  const int o = (tn << 4) + (lane & 15);
  wt[t] = (k < C) ? (h16)W[(size_t)o * C + k] : (h16)0.0f;
}

__global__ void prep_affine_kernel(const float* g, const float* be, const float* mu,
                                   const float* var, const float* bias,
                                   float* __restrict__ scale, float* __restrict__ shift, int O) {
  const int o = blockIdx.x * blockDim.x + threadIdx.x;
  if (o >= O) return;
  const float s = g ? g[o] * rsqrtf(var[o] + 1e-5f) : 1.0f;
  const float b = bias ? bias[o] : 0.0f;
  const float m = mu ? mu[o] : 0.0f;
  scale[o] = s;
  shift[o] = (b - m) * s + (be ? be[o] : 0.0f);
}

// ---------------------------------------------------------------------------
// WMMA GEMM: one wave = one 16x64 tile of D = A(MxK) * B(KxO), K%32==0, O%64==0.
// A: row-major f16 (two b128 loads / k-step). B: fragment-packed (see pack_w),
// so each of the 4 B fragments is one contiguous 32-byte load.
// Inner loop: 2 A-b128 + 8 B-b128 + 4 v_wmma. Epilogue: BN scale/shift (+ReLU).
// ---------------------------------------------------------------------------
__global__ void gemm_wmma_kernel(const h16* __restrict__ A, const h16* __restrict__ Bp,
                                 const float* __restrict__ scale, const float* __restrict__ shift,
                                 h16* __restrict__ outH, float* __restrict__ outF,
                                 int M, int K, int O, int relu) {
  const int lane = threadIdx.x & 31;
  const int wave = threadIdx.x >> 5;
  const int tilesM  = M >> 4;
  const int tilesO4 = O >> 6;
  const int t = blockIdx.x * 8 + wave;
  if (t >= tilesM * tilesO4) return;          // wave-uniform: EXEC stays all-1s
  const int tm  = t % tilesM;
  const int to4 = t / tilesM;
  const int m   = lane & 15;
  const int kbA = (lane >> 4) << 4;           // 0 or 16
  const int KB  = K >> 5;
  const size_t nstride = (size_t)KB << 9;     // elements per 16-col panel
  const h16* arow  = A + (size_t)(tm * 16 + m) * K + kbA;
  const h16* bbase = Bp + (size_t)(to4 * 4) * nstride + (size_t)lane * 16;
  v8f acc0 = {}, acc1 = {}, acc2 = {}, acc3 = {};
  for (int kb = 0; kb < KB; ++kb) {
    const int k0 = kb << 5;
    __builtin_prefetch(arow + k0 + 32, 0, 0);   // -> global_prefetch_b8
    union { v16h v; struct { v8h lo; v8h hi; } s; } av;
    av.s.lo = *(const v8h*)(arow + k0);         // K = kbA .. kbA+7
    av.s.hi = *(const v8h*)(arow + k0 + 8);     // K = kbA+8 .. kbA+15
    const h16* b0 = bbase + ((size_t)kb << 9);
    const v16h bv0 = *(const v16h*)(b0);
    const v16h bv1 = *(const v16h*)(b0 + nstride);
    const v16h bv2 = *(const v16h*)(b0 + 2 * nstride);
    const v16h bv3 = *(const v16h*)(b0 + 3 * nstride);
    acc0 = __builtin_amdgcn_wmma_f32_16x16x32_f16(false, av.v, false, bv0, (short)0, acc0, false, false);
    acc1 = __builtin_amdgcn_wmma_f32_16x16x32_f16(false, av.v, false, bv1, (short)0, acc1, false, false);
    acc2 = __builtin_amdgcn_wmma_f32_16x16x32_f16(false, av.v, false, bv2, (short)0, acc2, false, false);
    acc3 = __builtin_amdgcn_wmma_f32_16x16x32_f16(false, av.v, false, bv3, (short)0, acc3, false, false);
  }
  const int n     = lane & 15;
  const int mbase = tm * 16 + ((lane >> 4) << 3);
  v8f accs[4] = {acc0, acc1, acc2, acc3};
#pragma unroll
  for (int q = 0; q < 4; ++q) {
    const int oc = (to4 * 4 + q) * 16 + n;
    const float sc = scale[oc], sh = shift[oc];
#pragma unroll
    for (int r = 0; r < 8; ++r) {
      float y = accs[q][r] * sc + sh;
      if (relu) y = fmaxf(y, 0.0f);
      const size_t off = (size_t)(mbase + r) * O + oc;
      if (outH) outH[off] = (h16)y;
      if (outF) outF[off] = y;
    }
  }
}

// ---------------------------------------------------------------------------
// Host orchestration
// ---------------------------------------------------------------------------
extern "C" void kernel_launch(void* const* d_in, const int* in_sizes, int n_in,
                              void* d_out, int out_size, void* d_ws, size_t ws_size,
                              hipStream_t stream) {
  (void)in_sizes; (void)n_in; (void)out_size; (void)ws_size;
  constexpr int Bb = 16, N0 = 4096, S1 = 512, NS1 = 32, S2 = 128, NS2 = 64, N2 = 128;

  int gi = 0;
  const float* xyz = (const float*)d_in[gi++];

  struct LP { const float *W, *b, *g, *be, *mu, *var; int C, Cp, O; };
  LP L[17];
  auto grab = [&](int C, int Cp, int O, bool bn) {
    LP l;
    l.W = (const float*)d_in[gi++]; l.b = (const float*)d_in[gi++];
    if (bn) {
      l.g  = (const float*)d_in[gi++]; l.be  = (const float*)d_in[gi++];
      l.mu = (const float*)d_in[gi++]; l.var = (const float*)d_in[gi++];
    } else { l.g = nullptr; l.be = nullptr; l.mu = nullptr; l.var = nullptr; }
    l.C = C; l.Cp = Cp; l.O = O; return l;
  };
  L[0]  = grab(6,    32,   64,  true);  L[1]  = grab(64,  64,  64,   true);  L[2]  = grab(64, 64, 128, true);
  L[3]  = grab(131,  160,  128, true);  L[4]  = grab(128, 128, 128,  true);  L[5]  = grab(128,128, 256, true);
  L[6]  = grab(259,  288,  256, true);  L[7]  = grab(256, 256, 512,  true);  L[8]  = grab(512,512,1024, true);
  L[9]  = grab(1280, 1280, 256, true);  L[10] = grab(256, 256, 256,  true);
  L[11] = grab(384,  384,  256, true);  L[12] = grab(256, 256, 128,  true);
  L[13] = grab(134,  160,  128, true);  L[14] = grab(128, 128, 128,  true);  L[15] = grab(128,128, 128, true);
  L[16] = grab(128,  128,  256, false);

  char* wsp = (char*)d_ws;
  size_t off = 0;
  auto alloc = [&](size_t bytes) -> void* {
    size_t a = (off + 255) & ~(size_t)255;
    void* p = wsp + a; off = a + bytes; return p;
  };

  h16*   wt[17]; float* scl[17]; float* shf[17];
  for (int i = 0; i < 17; ++i) {
    wt[i]  = (h16*)alloc((size_t)L[i].Cp * L[i].O * 2);
    scl[i] = (float*)alloc((size_t)L[i].O * 4);
    shf[i] = (float*)alloc((size_t)L[i].O * 4);
  }
  h16* arena0 = (h16*)alloc((size_t)131072 * 160 * 2);   // 40 MB max
  h16* arena1 = (h16*)alloc((size_t)262144 * 128 * 2);   // 64 MB max
  int*   fps1  = (int*)alloc((size_t)Bb * S1 * 4);
  int*   fps2  = (int*)alloc((size_t)Bb * S2 * 4);
  float* nx1   = (float*)alloc((size_t)Bb * S1 * 3 * 4);
  float* nx2   = (float*)alloc((size_t)Bb * S2 * 3 * 4);
  int*   bidx1 = (int*)alloc((size_t)Bb * S1 * NS1 * 4);
  int*   bidx2 = (int*)alloc((size_t)Bb * S2 * NS2 * 4);
  h16*   l1p   = (h16*)alloc((size_t)Bb * S1 * 128 * 2);
  h16*   l2p   = (h16*)alloc((size_t)Bb * S2 * 256 * 2);
  h16*   l3p   = (h16*)alloc((size_t)Bb * 1024 * 2);
  h16*   p2    = (h16*)alloc((size_t)Bb * S2 * 256 * 2);
  h16*   p1    = (h16*)alloc((size_t)Bb * S1 * 128 * 2);
  h16*   p0    = (h16*)alloc((size_t)Bb * N0 * 128 * 2);
  int*   kidx2 = (int*)alloc((size_t)Bb * S1 * 3 * 4);
  float* kw2   = (float*)alloc((size_t)Bb * S1 * 3 * 4);
  int*   kidx1 = (int*)alloc((size_t)Bb * N0 * 3 * 4);
  float* kw1   = (float*)alloc((size_t)Bb * N0 * 3 * 4);

  auto blk = [](int tot) { return dim3((tot + 255) / 256); };
  auto gemm = [&](const h16* Ain, int li, int M, int Kp, int O, int relu,
                  h16* oh, float* of) {
    const int tiles = (M / 16) * (O / 64);
    gemm_wmma_kernel<<<dim3((tiles + 7) / 8), dim3(256), 0, stream>>>(
        Ain, wt[li], scl[li], shf[li], oh, of, M, Kp, O, relu);
  };

  // weight / BN folding prep
  for (int i = 0; i < 17; ++i) {
    const int nW = L[i].Cp * L[i].O;
    pack_w_kernel<<<blk(nW), dim3(256), 0, stream>>>(L[i].W, wt[i], L[i].C, L[i].Cp, L[i].O, nW);
    prep_affine_kernel<<<blk(L[i].O), dim3(256), 0, stream>>>(
        L[i].g, L[i].be, L[i].mu, L[i].var, L[i].b, scl[i], shf[i], L[i].O);
  }

  int tot;
  // ---- SA1 ----
  fps_kernel<<<dim3(Bb), dim3(256), 0, stream>>>(xyz, N0, S1, fps1);
  tot = Bb * S1;
  ball_kernel<<<blk(tot), dim3(256), 0, stream>>>(xyz, N0, fps1, S1, 0.04f, NS1, bidx1, nx1, tot);
  tot = Bb * S1 * NS1 * 32;
  group_sa1_kernel<<<blk(tot), dim3(256), 0, stream>>>(xyz, nx1, bidx1, arena0, tot, S1, NS1, N0, 32);
  gemm(arena0, 0, Bb * S1 * NS1, 32, 64, 1, arena1, nullptr);
  gemm(arena1, 1, Bb * S1 * NS1, 64, 64, 1, arena0, nullptr);
  gemm(arena0, 2, Bb * S1 * NS1, 64, 128, 1, arena1, nullptr);
  tot = Bb * S1 * 128;
  maxpool_kernel<<<blk(tot), dim3(256), 0, stream>>>(arena1, l1p, tot, NS1, 128);

  // ---- SA2 ----
  fps_kernel<<<dim3(Bb), dim3(256), 0, stream>>>(nx1, S1, S2, fps2);
  tot = Bb * S2;
  ball_kernel<<<blk(tot), dim3(256), 0, stream>>>(nx1, S1, fps2, S2, 0.16f, NS2, bidx2, nx2, tot);
  tot = Bb * S2 * NS2 * 160;
  group_sa2_kernel<<<blk(tot), dim3(256), 0, stream>>>(nx1, l1p, nx2, bidx2, arena0, tot, S2, NS2, S1, 128, 160);
  gemm(arena0, 3, Bb * S2 * NS2, 160, 128, 1, arena1, nullptr);
  gemm(arena1, 4, Bb * S2 * NS2, 128, 128, 1, arena0, nullptr);
  gemm(arena0, 5, Bb * S2 * NS2, 128, 256, 1, arena1, nullptr);
  tot = Bb * S2 * 256;
  maxpool_kernel<<<blk(tot), dim3(256), 0, stream>>>(arena1, l2p, tot, NS2, 256);

  // ---- SA3 (group_all) ----
  tot = Bb * N2 * 288;
  group_all_kernel<<<blk(tot), dim3(256), 0, stream>>>(nx2, l2p, arena0, tot, 256, 288);
  gemm(arena0, 6, Bb * N2, 288, 256, 1, arena1, nullptr);
  gemm(arena1, 7, Bb * N2, 256, 512, 1, arena0, nullptr);
  gemm(arena0, 8, Bb * N2, 512, 1024, 1, arena1, nullptr);
  tot = Bb * 1024;
  maxpool_kernel<<<blk(tot), dim3(256), 0, stream>>>(arena1, l3p, tot, N2, 1024);

  // ---- FP3 (S==1 broadcast) ----
  tot = Bb * N2 * 1280;
  concat_fp3_kernel<<<blk(tot), dim3(256), 0, stream>>>(l2p, l3p, arena0, tot, 256, 1024, N2);
  gemm(arena0, 9, Bb * N2, 1280, 256, 1, arena1, nullptr);
  gemm(arena1, 10, Bb * N2, 256, 256, 1, p2, nullptr);

  // ---- FP2 ----
  tot = Bb * S1;
  knn3_kernel<<<blk(tot), dim3(256), 0, stream>>>(nx1, nx2, tot, S1, S2, kidx2, kw2);
  tot = Bb * S1 * 384;
  interp_fp2_kernel<<<blk(tot), dim3(256), 0, stream>>>(l1p, p2, kidx2, kw2, arena0, tot, S1, S2, 128, 256);
  gemm(arena0, 11, Bb * S1, 384, 256, 1, arena1, nullptr);
  gemm(arena1, 12, Bb * S1, 256, 128, 1, p1, nullptr);

  // ---- FP1 ----
  tot = Bb * N0;
  knn3_kernel<<<blk(tot), dim3(256), 0, stream>>>(xyz, nx1, tot, N0, S1, kidx1, kw1);
  tot = Bb * N0 * 160;
  interp_fp1_kernel<<<blk(tot), dim3(256), 0, stream>>>(xyz, p1, kidx1, kw1, arena0, tot, N0, S1, 128, 160);
  gemm(arena0, 13, Bb * N0, 160, 128, 1, arena1, nullptr);
  gemm(arena1, 14, Bb * N0, 128, 128, 1, arena0, nullptr);
  gemm(arena0, 15, Bb * N0, 128, 128, 1, p0, nullptr);

  // ---- conv1: f32 output straight to d_out ----
  gemm(p0, 16, Bb * N0, 128, 256, 0, nullptr, (float*)d_out);
}
